// SpatialLocalAwareAttention_1632087573310
// MI455X (gfx1250) — matrix operand order repaired and checked by reading, not compiled, not run on hardware
//
#include <hip/hip_runtime.h>
#include <hip/hip_bf16.h>

// ---------------------------------------------------------------------------
// SpatialLocalAwareAttention for MI455X (gfx1250, wave32, WMMA bf16 path)
//   B=2, T=4, H=W=32  -> BT=8, N=1024, D=256, heads=8, dh=32
// Pipeline: LN -> bf16 ; Wconv -> bf16 ; 5x proj GEMM (WMMA) ;
//           flash attention w/ 3x3 local branch (WMMA) ; out GEMM (WMMA)
// ---------------------------------------------------------------------------

typedef __attribute__((ext_vector_type(16))) __bf16 v16bf;
typedef __attribute__((ext_vector_type(8)))  __bf16 v8bf;
typedef __attribute__((ext_vector_type(8)))  float  v8f;

union AFrag   { v16bf v; v8bf h[2]; };
union BF16x16 { v8bf h[2]; __bf16 e[16]; };

__device__ __forceinline__ v8f wmma_bf16(v16bf a, v16bf b, v8f c) {
  // D = A(16x32 bf16) * B(32x16 bf16) + C(16x16 f32)
  return __builtin_amdgcn_wmma_f32_16x16x32_bf16(false, a, false, b,
                                                 (short)0, c, false, false);
}

#define NTOK   8192      // BT * N
#define NIMG   1024      // H*W
#define DMODEL 256
#define NHEADS 8
#define DHEAD  32
#define NBH    64        // BT * heads
#define WELEM  65536     // 256*256

// ---------------------------------------------------------------------------
// 1) LayerNorm over D for each of 8192 tokens; emit bf16 xn[token][d]
//    x layout: (B, D, T, H, W); token order: (b, t, img)
// ---------------------------------------------------------------------------
__global__ __launch_bounds__(256) void ln_kernel(const float* __restrict__ x,
                                                 const float* __restrict__ gamma,
                                                 const float* __restrict__ beta,
                                                 __bf16* __restrict__ xn) {
  const int lane  = threadIdx.x & 31;
  const int wave  = threadIdx.x >> 5;
  const int token = blockIdx.x * 8 + wave;        // grid.x = 1024
  const int bt  = token >> 10, img = token & 1023;
  const int b   = bt >> 2,     t   = bt & 3;
  const float* xb = x + (size_t)b * DMODEL * 4 * NIMG + (size_t)t * NIMG + img;
  const int d0 = lane * 8;                        // 8 channels per lane
  float vals[8];
  float s = 0.f, s2 = 0.f;
#pragma unroll
  for (int u = 0; u < 8; ++u) {
    float v = xb[(size_t)(d0 + u) * 4096];        // stride T*H*W between d's
    vals[u] = v;
    s += v; s2 = fmaf(v, v, s2);
  }
#pragma unroll
  for (int m = 1; m < 32; m <<= 1) {
    s  += __shfl_xor(s, m);
    s2 += __shfl_xor(s2, m);
  }
  const float mean = s * (1.f / 256.f);
  const float var  = fmaxf(s2 * (1.f / 256.f) - mean * mean, 0.f);
  const float rstd = rsqrtf(var + 1e-6f);
  BF16x16 o;
#pragma unroll
  for (int u = 0; u < 8; ++u)
    o.e[u] = (__bf16)((vals[u] - mean) * rstd * gamma[d0 + u] + beta[d0 + u]);
  *(v8bf*)(xn + (size_t)token * DMODEL + d0) = o.h[0];
}

// ---------------------------------------------------------------------------
// 2) fp32 -> bf16 weight conversion (6 matrices, 256x256 each, row-major)
//    order in dst: Wq, Wk, Wv, Wkl, Wvl, Wo
// ---------------------------------------------------------------------------
__global__ __launch_bounds__(256) void wconv_kernel(const float* __restrict__ Wq,
                                                    const float* __restrict__ Wk,
                                                    const float* __restrict__ Wv,
                                                    const float* __restrict__ Wkl,
                                                    const float* __restrict__ Wvl,
                                                    const float* __restrict__ Wo,
                                                    __bf16* __restrict__ dst) {
  const int z = blockIdx.y;
  const float* src;
  switch (z) {
    case 0: src = Wq;  break;
    case 1: src = Wk;  break;
    case 2: src = Wv;  break;
    case 3: src = Wkl; break;
    case 4: src = Wvl; break;
    default: src = Wo; break;
  }
  const int i = blockIdx.x * 256 + threadIdx.x;   // grid.x = 256
  dst[(size_t)z * WELEM + i] = (__bf16)src[i];
}

// ---------------------------------------------------------------------------
// 3) Fused projections: out = xn @ W^T  for W in {Wq,Wk,Wv,Wkl,Wvl}
//    One wave computes one 16x16 tile, K-loop = 8 WMMAs.
//    Writes q(+bias)/k/kl/vl as (bh, n, dh) bf16 and v transposed (bh, dh, n).
// ---------------------------------------------------------------------------
__global__ __launch_bounds__(32) void proj_kernel(const __bf16* __restrict__ xn,
                                                  const __bf16* __restrict__ wbf,
                                                  const float*  __restrict__ qbias,
                                                  __bf16* __restrict__ qb,
                                                  __bf16* __restrict__ kb,
                                                  __bf16* __restrict__ vtb,
                                                  __bf16* __restrict__ klb,
                                                  __bf16* __restrict__ vlb) {
  const int lane = threadIdx.x, ln = lane & 15, lg = lane >> 4;
  const int m0 = blockIdx.x * 16;                 // token tile   (512 tiles)
  const int n0 = blockIdx.y * 16;                 // channel tile (16 tiles)
  const int mat = blockIdx.z;                     // 0..4
  const __bf16* W    = wbf + (size_t)mat * WELEM;
  const __bf16* arow = xn + (size_t)(m0 + ln) * DMODEL;
  const __bf16* brow = W  + (size_t)(n0 + ln) * DMODEL;
  v8f acc = {};
#pragma unroll
  for (int k0 = 0; k0 < DMODEL; k0 += 32) {
    AFrag a;
    a.h[0] = *(const v8bf*)(arow + k0 + 16 * lg);       // K = 16g + e     (e 0..7)
    a.h[1] = *(const v8bf*)(arow + k0 + 16 * lg + 8);   // K = 16g + e     (e 8..15)
    v16bf b = *(const v16bf*)(brow + k0 + 16 * lg);     // K = 16g + e
    acc = wmma_bf16(a.v, b, acc);
  }
  const int d = n0 + ln, head = d >> 5, dh = d & 31;
#pragma unroll
  for (int r = 0; r < 8; ++r) {
    const int token = m0 + r + 8 * lg;
    const int bt = token >> 10, img = token & 1023;
    const size_t bh = (size_t)(bt * NHEADS + head);
    const float v = acc[r];
    switch (mat) {
      case 0: qb[(bh * NIMG + img) * DHEAD + dh] =
                  (__bf16)(v + qbias[head * DHEAD + dh]);            break;
      case 1: kb [(bh * NIMG + img) * DHEAD + dh] = (__bf16)v;       break;
      case 2: vtb[(bh * DHEAD + dh) * NIMG + img] = (__bf16)v;       break; // transposed
      case 3: klb[(bh * NIMG + img) * DHEAD + dh] = (__bf16)v;       break;
      default: vlb[(bh * NIMG + img) * DHEAD + dh] = (__bf16)v;      break;
    }
  }
}

// ---------------------------------------------------------------------------
// 4) Flash attention + 3x3 local branch.
//    grid = (64 query tiles, 64 bh); one wave per block.
// ---------------------------------------------------------------------------
__global__ __launch_bounds__(32) void attn_kernel(const __bf16* __restrict__ q,
                                                  const __bf16* __restrict__ k,
                                                  const __bf16* __restrict__ vt,
                                                  const __bf16* __restrict__ kl,
                                                  const __bf16* __restrict__ vl,
                                                  __bf16* __restrict__ Y) {
  __shared__ __align__(16) __bf16 pexp[16][32];   // exp(S) staging, C->A relayout
  __shared__ float og[16][32];
  __shared__ float mgS[16], lgS[16];

  const int lane = threadIdx.x, ln = lane & 15, lg = lane >> 4;
  const int bh   = blockIdx.y;
  const int bt   = bh >> 3, head = bh & 7;
  const int qt   = blockIdx.x;
  const float factor = 0.1767766952966369f;       // 1/sqrt(32)

  // Q A-fragment: rows = 16 queries of this tile, K = dh (32)
  const __bf16* qrow = q + ((size_t)bh * NIMG + qt * 16 + ln) * DHEAD;
  AFrag aq;
  aq.h[0] = *(const v8bf*)(qrow + 16 * lg);
  aq.h[1] = *(const v8bf*)(qrow + 16 * lg + 8);

  const __bf16* kbase = k  + (size_t)bh * NIMG * DHEAD;
  const __bf16* vbase = vt + (size_t)bh * DHEAD * NIMG;

  v8f o0 = {}, o1 = {};                           // O tiles: dh 0-15, dh 16-31
  float mrow[8], lrow[8];
#pragma unroll
  for (int r = 0; r < 8; ++r) { mrow[r] = -1e30f; lrow[r] = 0.f; }

  const v8f zero = {};
  for (int kt = 0; kt < 32; ++kt) {               // 32 keys per iteration
    const int key0 = kt * 32;
    // K B-fragments: B[kk][n] = K[key][kk] -> contiguous 32B per lane
    v16bf bk0 = *(const v16bf*)(kbase + (size_t)(key0 + ln) * DHEAD + 16 * lg);
    v16bf bk1 = *(const v16bf*)(kbase + (size_t)(key0 + 16 + ln) * DHEAD + 16 * lg);
    if (kt < 31) {                                // gfx1250 prefetch of next tile
      __builtin_prefetch(kbase + (size_t)(key0 + 32 + ln) * DHEAD, 0, 1);
      __builtin_prefetch(vbase + (size_t)ln * NIMG + key0 + 32, 0, 1);
    }
    v8f s0 = wmma_bf16(aq.v, bk0, zero);
    v8f s1 = wmma_bf16(aq.v, bk1, zero);
#pragma unroll
    for (int r = 0; r < 8; ++r) {                 // online softmax per row r+8g
      const float a = s0[r] * factor, b = s1[r] * factor;
      float t = fmaxf(a, b);
      t = fmaxf(t, __shfl_xor(t, 1));
      t = fmaxf(t, __shfl_xor(t, 2));
      t = fmaxf(t, __shfl_xor(t, 4));
      t = fmaxf(t, __shfl_xor(t, 8));
      const float mnew  = fmaxf(mrow[r], t);
      const float scale = __expf(mrow[r] - mnew);
      const float p0 = __expf(a - mnew);
      const float p1 = __expf(b - mnew);
      float rs = p0 + p1;
      rs += __shfl_xor(rs, 1);
      rs += __shfl_xor(rs, 2);
      rs += __shfl_xor(rs, 4);
      rs += __shfl_xor(rs, 8);
      lrow[r] = lrow[r] * scale + rs;
      mrow[r] = mnew;
      o0[r] *= scale; o1[r] *= scale;
      pexp[r + 8 * lg][ln]      = (__bf16)p0;
      pexp[r + 8 * lg][16 + ln] = (__bf16)p1;
    }
    __syncthreads();
    AFrag ap;                                     // P as A-fragment (K = 32 keys)
    ap.h[0] = *(const v8bf*)(&pexp[ln][16 * lg]);
    ap.h[1] = *(const v8bf*)(&pexp[ln][16 * lg + 8]);
    // V B-fragments from transposed V: B[kk][n] = Vt[dh][key] contiguous
    v16bf bv0 = *(const v16bf*)(vbase + (size_t)ln * NIMG + key0 + 16 * lg);
    v16bf bv1 = *(const v16bf*)(vbase + (size_t)(16 + ln) * NIMG + key0 + 16 * lg);
    o0 = wmma_bf16(ap.v, bv0, o0);
    o1 = wmma_bf16(ap.v, bv1, o1);
    __syncthreads();
  }

  // Dump global-branch state to LDS for re-layout.
#pragma unroll
  for (int r = 0; r < 8; ++r) {
    og[r + 8 * lg][ln]      = o0[r];
    og[r + 8 * lg][16 + ln] = o1[r];
  }
  if (ln == 0) {
#pragma unroll
    for (int r = 0; r < 8; ++r) { mgS[r + 8 * lg] = mrow[r]; lgS[r + 8 * lg] = lrow[r]; }
  }
  __syncthreads();

  // ---- local 3x3 branch: lane = (row m = ln, dh half hf = lg) ----
  const int m = ln, hf = lg;
  const int tok = qt * 16 + m;
  const int pi = tok >> 5, pj = tok & 31;
  float qv[16];
  {
    BF16x16 u;
    const __bf16* p = q + ((size_t)bh * NIMG + tok) * DHEAD + 16 * hf;
    u.h[0] = *(const v8bf*)p; u.h[1] = *(const v8bf*)(p + 8);
#pragma unroll
    for (int d = 0; d < 16; ++d) qv[d] = (float)u.e[d];
  }
  float sc[9]; int nbs[9]; int cnt = 0;
  float mloc = -1e30f;
#pragma unroll
  for (int di = -1; di <= 1; ++di)
#pragma unroll
    for (int dj = -1; dj <= 1; ++dj) {
      const int ni = pi + di, nj = pj + dj;
      if (ni < 0 || ni >= 32 || nj < 0 || nj >= 32) continue;
      const int nb = ni * 32 + nj;
      BF16x16 u;
      const __bf16* p = kl + ((size_t)bh * NIMG + nb) * DHEAD + 16 * hf;
      u.h[0] = *(const v8bf*)p; u.h[1] = *(const v8bf*)(p + 8);
      float part = 0.f;
#pragma unroll
      for (int d = 0; d < 16; ++d) part = fmaf(qv[d], (float)u.e[d], part);
      const float full = (part + __shfl_xor(part, 16)) * factor;
      sc[cnt] = full; nbs[cnt] = nb; ++cnt;
      mloc = fmaxf(mloc, full);
    }
  float lloc = 0.f;
  float oloc[16];
#pragma unroll
  for (int d = 0; d < 16; ++d) oloc[d] = 0.f;
  for (int i = 0; i < cnt; ++i) {
    const float p = __expf(sc[i] - mloc);
    lloc += p;
    BF16x16 u;
    const __bf16* vp = vl + ((size_t)bh * NIMG + nbs[i]) * DHEAD + 16 * hf;
    u.h[0] = *(const v8bf*)vp; u.h[1] = *(const v8bf*)(vp + 8);
#pragma unroll
    for (int d = 0; d < 16; ++d) oloc[d] = fmaf(p, (float)u.e[d], oloc[d]);
  }
  // ---- merge global & local flash states, write Y row-major bf16 ----
  const float mgv = mgS[m], lgv = lgS[m];
  const float mf = fmaxf(mgv, mloc);
  const float eg = __expf(mgv - mf), el = __expf(mloc - mf);
  const float inv = 1.f / (lgv * eg + lloc * el);
  BF16x16 outv;
#pragma unroll
  for (int d = 0; d < 16; ++d)
    outv.e[d] = (__bf16)((og[m][16 * hf + d] * eg + oloc[d] * el) * inv);
  __bf16* yp = Y + ((size_t)bt * NIMG + tok) * DMODEL + head * DHEAD + 16 * hf;
  *(v8bf*)yp = outv.h[0];
  *(v8bf*)(yp + 8) = outv.h[1];
}

// ---------------------------------------------------------------------------
// 5) Output GEMM: out = Y @ Wo^T, scattered into (B, D, T, H, W) fp32.
// ---------------------------------------------------------------------------
__global__ __launch_bounds__(32) void outgemm_kernel(const __bf16* __restrict__ Y,
                                                     const __bf16* __restrict__ Wo,
                                                     float* __restrict__ out) {
  const int lane = threadIdx.x, ln = lane & 15, lg = lane >> 4;
  const int m0 = blockIdx.x * 16, n0 = blockIdx.y * 16;
  const __bf16* arow = Y  + (size_t)(m0 + ln) * DMODEL;
  const __bf16* brow = Wo + (size_t)(n0 + ln) * DMODEL;
  v8f acc = {};
#pragma unroll
  for (int k0 = 0; k0 < DMODEL; k0 += 32) {
    AFrag a;
    a.h[0] = *(const v8bf*)(arow + k0 + 16 * lg);
    a.h[1] = *(const v8bf*)(arow + k0 + 16 * lg + 8);
    v16bf b = *(const v16bf*)(brow + k0 + 16 * lg);
    acc = wmma_bf16(a.v, b, acc);
  }
  const int d = n0 + ln;
#pragma unroll
  for (int r = 0; r < 8; ++r) {
    const int token = m0 + r + 8 * lg;
    const int bt = token >> 10, img = token & 1023;
    const int b = bt >> 2, t = bt & 3;
    out[(((size_t)b * DMODEL + d) * 4 + t) * NIMG + img] = acc[r];
  }
}

// ---------------------------------------------------------------------------
extern "C" void kernel_launch(void* const* d_in, const int* in_sizes, int n_in,
                              void* d_out, int out_size, void* d_ws, size_t ws_size,
                              hipStream_t stream) {
  const float* x     = (const float*)d_in[0];
  const float* gamma = (const float*)d_in[1];
  const float* beta  = (const float*)d_in[2];
  const float* Wq    = (const float*)d_in[3];
  const float* Wk    = (const float*)d_in[4];
  const float* Wv    = (const float*)d_in[5];
  const float* Wkl   = (const float*)d_in[6];
  const float* Wvl   = (const float*)d_in[7];
  const float* Wo    = (const float*)d_in[8];
  const float* qbias = (const float*)d_in[9];
  float* out = (float*)d_out;

  // Workspace layout (bf16 elements); total ~30 MB, fully L2-resident.
  __bf16* ws  = (__bf16*)d_ws;
  const size_t TOKD = (size_t)NTOK * DMODEL;      // 2,097,152
  __bf16* xn  = ws;
  __bf16* qb  = xn  + TOKD;
  __bf16* kb  = qb  + TOKD;
  __bf16* vtb = kb  + TOKD;
  __bf16* klb = vtb + TOKD;
  __bf16* vlb = klb + TOKD;
  __bf16* Yb  = vlb + TOKD;
  __bf16* wbf = Yb  + TOKD;                       // 6 * 65536

  wconv_kernel<<<dim3(WELEM / 256, 6), 256, 0, stream>>>(Wq, Wk, Wv, Wkl, Wvl, Wo, wbf);
  ln_kernel<<<NTOK / 8, 256, 0, stream>>>(x, gamma, beta, xn);
  proj_kernel<<<dim3(NTOK / 16, DMODEL / 16, 5), 32, 0, stream>>>(
      xn, wbf, qbias, qb, kb, vtb, klb, vlb);
  attn_kernel<<<dim3(NIMG / 16, NBH), 32, 0, stream>>>(qb, kb, vtb, klb, vlb, Yb);
  outgemm_kernel<<<dim3(NTOK / 16, DMODEL / 16), 32, 0, stream>>>(
      Yb, wbf + (size_t)5 * WELEM, out);
}
